// PointCloudEmbedding_67654324847148
// MI455X (gfx1250) — matrix operand order repaired and checked by reading, not compiled, not run on hardware
//
#include <hip/hip_runtime.h>
#include <stdint.h>

// ---------------------------------------------------------------------------
// PointNet-style embed + 3x (1x1 conv -> BN(train) -> ReLU) for MI455X/gfx1250
// wave32, WMMA f32_16x16x32_f16. Output-bandwidth-bound: 512MB fp32 out.
// Block-3 BN stats are computed in a discard pass (no y3 materialization),
// then the GEMM is recomputed with a fused BN+ReLU epilogue and non-temporal
// stores so the 512MB stream does not evict the L2-resident x2/W3 working set.
// ---------------------------------------------------------------------------

typedef _Float16 f16;
typedef __attribute__((ext_vector_type(16))) _Float16 v16h;
typedef __attribute__((ext_vector_type(8)))  _Float16 v8h;
typedef __attribute__((ext_vector_type(8)))  float    v8f;
typedef __attribute__((ext_vector_type(4)))  float    v4f;

#define NPTS   131072      // B*N = 4*32768
#define MTOT   131072
#define NBQ    33          // NB
#define NB3    35937       // 33^3
#define NBINS  143748      // 4 * 33^3
#define NSTAT  2432        // 2*(64+128+1024)
#define EPSBN  1e-5f

// order-preserving float <-> uint map (for atomic min/max on floats)
__device__ __forceinline__ uint32_t fmap(float f) {
  uint32_t u = __float_as_uint(f);
  return (u & 0x80000000u) ? ~u : (u | 0x80000000u);
}
__device__ __forceinline__ float funmap(uint32_t u) {
  u = (u & 0x80000000u) ? (u & 0x7fffffffu) : ~u;
  return __uint_as_float(u);
}

// ------------------------- init (zero counts/stats, init minmax) -----------
__global__ void init_kernel(uint32_t* __restrict__ counts,
                            float* __restrict__ stats,
                            uint32_t* __restrict__ mm) {
  int i = blockIdx.x * 256 + threadIdx.x;
  if (i < NBINS) {
    counts[i] = 0u;
  } else if (i < NBINS + NSTAT) {
    stats[i - NBINS] = 0.0f;
  } else if (i < NBINS + NSTAT + 6) {
    int k = i - NBINS - NSTAT;
    mm[k] = (k < 3) ? 0xFFFFFFFFu : 0u;   // mins -> +inf mapped, maxs -> -inf mapped
  }
}

// ------------------------- global per-axis min/max -------------------------
__global__ void minmax_kernel(const float* __restrict__ pt, uint32_t* __restrict__ mm) {
  const int p = blockIdx.x * 256 + threadIdx.x;
  uint32_t lo[3], hi[3];
#pragma unroll
  for (int a = 0; a < 3; ++a) {
    uint32_t u = fmap(pt[p * 3 + a]);
    lo[a] = u; hi[a] = u;
  }
#pragma unroll
  for (int off = 16; off > 0; off >>= 1) {
#pragma unroll
    for (int a = 0; a < 3; ++a) {
      uint32_t l = (uint32_t)__shfl_xor((int)lo[a], off);
      uint32_t h = (uint32_t)__shfl_xor((int)hi[a], off);
      lo[a] = (l < lo[a]) ? l : lo[a];
      hi[a] = (h > hi[a]) ? h : hi[a];
    }
  }
  if ((threadIdx.x & 31) == 0) {
#pragma unroll
    for (int a = 0; a < 3; ++a) {
      atomicMin(&mm[a], lo[a]);
      atomicMax(&mm[3 + a], hi[a]);
    }
  }
}

// voxel id for point p; optionally return float voxel coords
__device__ __forceinline__ int voxel_flat(const float* __restrict__ pt,
                                          const uint32_t* __restrict__ mm,
                                          int p, float* vf_out) {
  int vi[3];
#pragma unroll
  for (int a = 0; a < 3; ++a) {
    float mn = funmap(mm[a]);
    float mx = funmap(mm[3 + a]);
    float cell = (mx - mn) * (1.0f / 32.0f);
    float x = pt[p * 3 + a];
    float v = floorf((x - mn) / cell);
    v = fminf(fmaxf(v, 0.0f), 32.0f);
    if (vf_out) vf_out[a] = v;
    vi[a] = (int)v;
  }
  int b = p >> 15;    // N = 32768
  return b * NB3 + (vi[0] * NBQ + vi[1]) * NBQ + vi[2];
}

// ------------------------- voxel occupancy histogram -----------------------
__global__ void hist_kernel(const float* __restrict__ pt,
                            const uint32_t* __restrict__ mm,
                            uint32_t* __restrict__ counts) {
  const int p = blockIdx.x * 256 + threadIdx.x;
  atomicAdd(&counts[voxel_flat(pt, mm, p, nullptr)], 1u);
}

// ------------------------- features [M, 32] f16 (K padded 4 -> 32) ---------
__global__ void embed_kernel(const float* __restrict__ pt,
                             const uint32_t* __restrict__ mm,
                             const uint32_t* __restrict__ counts,
                             f16* __restrict__ X0) {
  const int p = blockIdx.x * 256 + threadIdx.x;
  float vf[3];
  int flat = voxel_flat(pt, mm, p, vf);
  float cnt = (float)counts[flat];
  v8h r0 = {};
  r0[0] = (f16)vf[0]; r0[1] = (f16)vf[1]; r0[2] = (f16)vf[2]; r0[3] = (f16)cnt;
  v8h z = {};
  v8h* dst = (v8h*)(X0 + (size_t)p * 32);
  dst[0] = r0; dst[1] = z; dst[2] = z; dst[3] = z;
}

// ------------------------- f32 weights -> f16, K-padded --------------------
__global__ void wconv_kernel(const float* __restrict__ W, f16* __restrict__ Wh,
                             int cout, int cins, int cind) {
  int i = blockIdx.x * 256 + threadIdx.x;
  if (i >= cout * cind) return;
  int o = i / cind, k = i % cind;
  Wh[i] = (k < cins) ? (f16)W[o * cins + k] : (f16)0.0f;
}

// ------------------------- BN stats -> scale/shift -------------------------
__global__ void finalize_kernel(const float* __restrict__ stats,
                                const float* __restrict__ g,
                                const float* __restrict__ beta,
                                float* __restrict__ scsh, int C) {
  int c = blockIdx.x * blockDim.x + threadIdx.x;
  if (c >= C) return;
  const float invM = 1.0f / (float)MTOT;
  float mean = stats[c] * invM;
  float var  = stats[C + c] * invM - mean * mean;
  float sc = g[c] * rsqrtf(var + EPSBN);
  scsh[c] = sc;
  scsh[C + c] = beta[c] - mean * sc;
}

// ------------------------- x = relu(scale*y + shift), f16 -> f16 -----------
__global__ void apply_kernel(const f16* __restrict__ Y, const float* __restrict__ scsh,
                             f16* __restrict__ X, int C, int total8) {
  int i = blockIdx.x * 256 + threadIdx.x;
  if (i >= total8) return;
  int cbase = (i * 8) % C;
  v8h y = *(const v8h*)(Y + (size_t)i * 8);
  v8h x;
#pragma unroll
  for (int r = 0; r < 8; ++r) {
    int c = cbase + r;
    float v = (float)y[r] * scsh[c] + scsh[C + c];
    x[r] = (f16)fmaxf(v, 0.0f);
  }
  *(v8h*)(X + (size_t)i * 8) = x;
}

// ---------------------------------------------------------------------------
// WMMA GEMM: Y[M,COUT] = A[M,CIN] * W[COUT,CIN]^T + bias
// MODE 0: store Y f16 + accumulate per-channel sum/sumsq
// MODE 1: stats only (no store)           -- block-3 pass 1
// MODE 2: Out = relu(scale*Y+shift) f32   -- block-3 pass 2 (NT stores)
// Workgroup: 256 threads = 8 waves; wave tile 32Mx64N (2x4 WMMA frags).
// ---------------------------------------------------------------------------
template<int CIN, int COUT, int MODE>
__global__ void __launch_bounds__(256)
gemm_kernel(const f16* __restrict__ A, const f16* __restrict__ W,
            const float* __restrict__ bias, f16* __restrict__ Y,
            float* __restrict__ stats, const float* __restrict__ scsh,
            float* __restrict__ Out) {
  constexpr int NWN = (COUT >= 128) ? 2 : 1;
  constexpr int NWM = 8 / NWN;
  constexpr int WGM = NWM * 32;
  constexpr int WGN = NWN * 64;

  const int tid  = threadIdx.x;
  const int lane = tid & 31;
  const int wave = tid >> 5;
  const int lrow = lane & 15;     // C/D column within frag; A/B row within frag
  const int lsel = lane >> 4;     // lane half selector
  const int mw = wave % NWM;
  const int nw = wave / NWM;
  const int m0 = blockIdx.x * WGM + mw * 32;
  const int n0 = blockIdx.y * WGN + nw * 64;

  v8f acc[2][4] = {};

  // A per-lane: 16 contiguous K at K = k+16*lsel (two 8-half chunks), matching B
  const f16* Ap = A + (size_t)(m0 + lrow) * CIN + lsel * 16;
  // B per-lane: 16 contiguous K at K = k+16*lsel, row = output channel
  const f16* Wp = W + (size_t)(n0 + lrow) * CIN + lsel * 16;

#pragma unroll
  for (int k = 0; k < CIN; k += 32) {
    v8h al0 = *(const v8h*)(Ap + k);
    v8h ah0 = *(const v8h*)(Ap + k + 8);
    v8h al1 = *(const v8h*)(Ap + 16 * CIN + k);
    v8h ah1 = *(const v8h*)(Ap + 16 * CIN + k + 8);
    v16h a0 = __builtin_shufflevector(al0, ah0, 0,1,2,3,4,5,6,7,8,9,10,11,12,13,14,15);
    v16h a1 = __builtin_shufflevector(al1, ah1, 0,1,2,3,4,5,6,7,8,9,10,11,12,13,14,15);
    v16h b0 = *(const v16h*)(Wp + k);
    v16h b1 = *(const v16h*)(Wp + 16 * CIN + k);
    v16h b2 = *(const v16h*)(Wp + 32 * CIN + k);
    v16h b3 = *(const v16h*)(Wp + 48 * CIN + k);
    acc[0][0] = __builtin_amdgcn_wmma_f32_16x16x32_f16(false, a0, false, b0, (short)0, acc[0][0], false, false);
    acc[0][1] = __builtin_amdgcn_wmma_f32_16x16x32_f16(false, a0, false, b1, (short)0, acc[0][1], false, false);
    acc[0][2] = __builtin_amdgcn_wmma_f32_16x16x32_f16(false, a0, false, b2, (short)0, acc[0][2], false, false);
    acc[0][3] = __builtin_amdgcn_wmma_f32_16x16x32_f16(false, a0, false, b3, (short)0, acc[0][3], false, false);
    acc[1][0] = __builtin_amdgcn_wmma_f32_16x16x32_f16(false, a1, false, b0, (short)0, acc[1][0], false, false);
    acc[1][1] = __builtin_amdgcn_wmma_f32_16x16x32_f16(false, a1, false, b1, (short)0, acc[1][1], false, false);
    acc[1][2] = __builtin_amdgcn_wmma_f32_16x16x32_f16(false, a1, false, b2, (short)0, acc[1][2], false, false);
    acc[1][3] = __builtin_amdgcn_wmma_f32_16x16x32_f16(false, a1, false, b3, (short)0, acc[1][3], false, false);
  }

  // bias (per D column: N = lane&15)
#pragma unroll
  for (int j = 0; j < 4; ++j) {
    const float bv = bias[n0 + j * 16 + lrow];
#pragma unroll
    for (int i = 0; i < 2; ++i)
#pragma unroll
      for (int r = 0; r < 8; ++r) acc[i][j][r] += bv;
  }

  // ---- per-channel sum / sumsq (for training-mode BN) ----
  if constexpr (MODE != 2) {
    __shared__ float s_sum[WGN];
    __shared__ float s_sq[WGN];
    if (tid < WGN) { s_sum[tid] = 0.0f; s_sq[tid] = 0.0f; }
    __syncthreads();
#pragma unroll
    for (int j = 0; j < 4; ++j) {
      float s = 0.0f, q = 0.0f;
#pragma unroll
      for (int i = 0; i < 2; ++i)
#pragma unroll
        for (int r = 0; r < 8; ++r) { float v = acc[i][j][r]; s += v; q += v * v; }
      s += __shfl_xor(s, 16);           // fold the two row-halves (same column)
      q += __shfl_xor(q, 16);
      if (lane < 16) {
        atomicAdd(&s_sum[nw * 64 + j * 16 + lane], s);   // ds_add_f32
        atomicAdd(&s_sq [nw * 64 + j * 16 + lane], q);
      }
    }
    __syncthreads();
    if (tid < WGN) {
      atomicAdd(&stats[blockIdx.y * WGN + tid], s_sum[tid]);
      atomicAdd(&stats[COUT + blockIdx.y * WGN + tid], s_sq[tid]);
    }
  }

  // ---- store Y as f16 via LDS repack (coalesced 16B stores, stay in L2) ----
  if constexpr (MODE == 0) {
    __shared__ __align__(16) f16 tile[WGM][WGN + 8];
#pragma unroll
    for (int j = 0; j < 4; ++j)
#pragma unroll
      for (int i = 0; i < 2; ++i)
#pragma unroll
        for (int r = 0; r < 8; ++r)
          tile[mw * 32 + i * 16 + lsel * 8 + r][nw * 64 + j * 16 + lrow] = (f16)acc[i][j][r];
    __syncthreads();
    constexpr int CPR = WGN / 8;
    for (int e = tid; e < WGM * CPR; e += 256) {
      const int r = e / CPR, c = e % CPR;
      v8h v = *(const v8h*)&tile[r][c * 8];
      *(v8h*)(Y + (size_t)(blockIdx.x * WGM + r) * COUT + blockIdx.y * WGN + c * 8) = v;
    }
  }

  // ---- fused BN+ReLU epilogue, f32 out, non-temporal streaming stores ----
  if constexpr (MODE == 2) {
    __shared__ __align__(16) float tf[WGM / 2][WGN + 4];
    float sc[4], sh[4];
#pragma unroll
    for (int j = 0; j < 4; ++j) {
      const int col = blockIdx.y * WGN + nw * 64 + j * 16 + lrow;
      sc[j] = scsh[col];
      sh[j] = scsh[COUT + col];
    }
    constexpr int HM = NWM / 2;
#pragma unroll
    for (int half = 0; half < 2; ++half) {
      if (mw / HM == half) {
#pragma unroll
        for (int j = 0; j < 4; ++j)
#pragma unroll
          for (int i = 0; i < 2; ++i)
#pragma unroll
            for (int r = 0; r < 8; ++r) {
              float v = fmaxf(acc[i][j][r] * sc[j] + sh[j], 0.0f);
              tf[mw * 32 + i * 16 + lsel * 8 + r - half * (WGM / 2)][nw * 64 + j * 16 + lrow] = v;
            }
      }
      __syncthreads();
      constexpr int CPR = WGN / 4;
      for (int e = tid; e < (WGM / 2) * CPR; e += 256) {
        const int r = e / CPR, c = e % CPR;
        v4f v = *(const v4f*)&tf[r][c * 4];
        // 512MB output stream: TH=NT so it does not rinse x2/W3 out of L2
        __builtin_nontemporal_store(
            v, (v4f*)(Out + (size_t)(blockIdx.x * WGM + half * (WGM / 2) + r) * COUT
                           + blockIdx.y * WGN + c * 4));
      }
      __syncthreads();
    }
  }
}

// ---------------------------------------------------------------------------
extern "C" void kernel_launch(void* const* d_in, const int* in_sizes, int n_in,
                              void* d_out, int out_size, void* d_ws, size_t ws_size,
                              hipStream_t stream) {
  const float* pt  = (const float*)d_in[0];
  const float* W1  = (const float*)d_in[1];
  const float* b1  = (const float*)d_in[2];
  const float* g1  = (const float*)d_in[3];
  const float* be1 = (const float*)d_in[4];
  const float* W2  = (const float*)d_in[5];
  const float* b2  = (const float*)d_in[6];
  const float* g2  = (const float*)d_in[7];
  const float* be2 = (const float*)d_in[8];
  const float* W3  = (const float*)d_in[9];
  const float* b3  = (const float*)d_in[10];
  const float* g3  = (const float*)d_in[11];
  const float* be3 = (const float*)d_in[12];
  float* out = (float*)d_out;

  char* ws = (char*)d_ws;
  uint32_t* mm    = (uint32_t*)(ws);                      // 6 mapped min/max
  float*    scsh  = (float*)(ws + 64);                    // 2432 scale/shift
  float*    stats = (float*)(ws + 16 * 1024);             // 2432 sums
  uint32_t* cnts  = (uint32_t*)(ws + 32 * 1024);          // 143748 histogram
  f16* Wh1 = (f16*)(ws + (1u << 20));                     // 64x32 f16
  f16* Wh2 = (f16*)(ws + (1u << 20) + (8u << 10));        // 128x64 f16
  f16* Wh3 = (f16*)(ws + (1u << 20) + (32u << 10));       // 1024x128 f16
  f16* X0  = (f16*)(ws + (2u  << 20));                    //  8 MB
  f16* Y1  = (f16*)(ws + (10u << 20));                    // 16 MB
  f16* X1  = (f16*)(ws + (26u << 20));                    // 16 MB
  f16* Y2  = (f16*)(ws + (42u << 20));                    // 32 MB
  f16* X2  = (f16*)(ws + (74u << 20));                    // 32 MB (total ~106 MB)

  init_kernel<<<(NBINS + NSTAT + 6 + 255) / 256, 256, 0, stream>>>(cnts, stats, mm);
  minmax_kernel<<<NPTS / 256, 256, 0, stream>>>(pt, mm);
  hist_kernel<<<NPTS / 256, 256, 0, stream>>>(pt, mm, cnts);
  wconv_kernel<<<(64 * 32 + 255) / 256, 256, 0, stream>>>(W1, Wh1, 64, 4, 32);
  wconv_kernel<<<(128 * 64 + 255) / 256, 256, 0, stream>>>(W2, Wh2, 128, 64, 64);
  wconv_kernel<<<(1024 * 128 + 255) / 256, 256, 0, stream>>>(W3, Wh3, 1024, 128, 128);
  embed_kernel<<<NPTS / 256, 256, 0, stream>>>(pt, mm, cnts, X0);

  // block 1: [M,32] x [64,32]^T  (K padded from 4)
  gemm_kernel<32, 64, 0><<<dim3(MTOT / 256, 1), 256, 0, stream>>>(
      X0, Wh1, b1, Y1, stats + 0, nullptr, nullptr);
  finalize_kernel<<<1, 64, 0, stream>>>(stats + 0, g1, be1, scsh + 0, 64);
  apply_kernel<<<(MTOT * 64 / 8) / 256, 256, 0, stream>>>(Y1, scsh + 0, X1, 64, MTOT * 64 / 8);

  // block 2: [M,64] x [128,64]^T
  gemm_kernel<64, 128, 0><<<dim3(MTOT / 128, 1), 256, 0, stream>>>(
      X1, Wh2, b2, Y2, stats + 128, nullptr, nullptr);
  finalize_kernel<<<1, 128, 0, stream>>>(stats + 128, g2, be2, scsh + 128, 128);
  apply_kernel<<<(MTOT * 128 / 8) / 256, 256, 0, stream>>>(Y2, scsh + 128, X2, 128, MTOT * 128 / 8);

  // block 3: [M,128] x [1024,128]^T -- pass 1 stats only, pass 2 fused BN+ReLU -> out
  gemm_kernel<128, 1024, 1><<<dim3(MTOT / 128, 8), 256, 0, stream>>>(
      X2, Wh3, b3, nullptr, stats + 384, nullptr, nullptr);
  finalize_kernel<<<4, 256, 0, stream>>>(stats + 384, g3, be3, scsh + 384, 1024);
  gemm_kernel<128, 1024, 2><<<dim3(MTOT / 128, 8), 256, 0, stream>>>(
      X2, Wh3, b3, nullptr, nullptr, scsh + 384, out);
}
